// GraphEncoder_20229295964929
// MI455X (gfx1250) — matrix operand order repaired and checked by reading, not compiled, not run on hardware
//
#include <hip/hip_runtime.h>

#define HEADS     4
#define HEAD_DIM  32
#define HID       128
#define NEG_SLOPE 0.2f

typedef __attribute__((ext_vector_type(16))) _Float16 v16h;
typedef __attribute__((ext_vector_type(8)))  float    v8f;

// ---------------------------------------------------------------------------
// Order-preserving float<->uint encoding so we can use u32 atomicMax for the
// segment-max (GLOBAL_ATOMIC_MAX_U32). u=0 acts as -inf sentinel; every node
// has a self loop so every segment sees at least one real value.
// ---------------------------------------------------------------------------
__device__ __forceinline__ unsigned enc_f32(float f) {
  int i = __float_as_int(f);
  unsigned u = (unsigned)i;
  return (i >= 0) ? (u | 0x80000000u) : ~u;
}
__device__ __forceinline__ float dec_f32(unsigned u) {
  return (u & 0x80000000u) ? __int_as_float((int)(u & 0x7fffffffu))
                           : __int_as_float((int)~u);
}

// ---------------------------------------------------------------------------
// f32 -> f16 convert (activations)
// ---------------------------------------------------------------------------
__global__ void cvt_x_f16(const float* __restrict__ in, _Float16* __restrict__ out, int n) {
  int i = blockIdx.x * blockDim.x + threadIdx.x;
  if (i < n) out[i] = (_Float16)in[i];
}

// ---------------------------------------------------------------------------
// WMMA GEMM, persistent blocks: C[N,128] = A[N,128](f16) x W[128,128](f32->f16)
//
// Each 256-thread block (8 wave32s) stages the whole weight matrix once into a
// 32 KB LDS tile, transposed to Wt[n][k] so B-fragments read contiguous K via
// ds_load_b128. Waves then loop over 16-row output stripes strided by the
// total wave count: 8 column tiles x 4 K-steps = 32 v_wmma per stripe.
// ---------------------------------------------------------------------------
__global__ void __launch_bounds__(256)
gat_gemm_wmma(const _Float16* __restrict__ A, const float* __restrict__ W,
              float* __restrict__ C, int nRowTiles) {
  __shared__ _Float16 sW[HID * HID];             // 32 KB of the 320 KB WGP LDS

  // cooperative transpose + convert: sW[n*128+k] = (f16)W[k*128+n]
  for (int idx = threadIdx.x; idx < HID * HID; idx += 256) {
    int n = idx >> 7, k = idx & 127;
    sW[idx] = (_Float16)W[k * HID + n];
  }
  __syncthreads();

  const int lane       = threadIdx.x & 31;
  const int rlo        = lane & 15;
  const int hiHalf     = lane >> 4;              // 0: lanes 0-15, 1: lanes 16-31
  const int globalWave = blockIdx.x * 8 + (threadIdx.x >> 5);
  const int totalWaves = gridDim.x * 8;

  for (int stripe = globalWave; stripe < nRowTiles; stripe += totalWaves) {
    // A fragment base: row = stripe*16 + rlo; lanes>=16 hold K offset +8
    const _Float16* arowBase = A + (size_t)(stripe * 16 + rlo) * HID + hiHalf * 8;

    v8f acc[8] = {};

#pragma unroll
    for (int kt = 0; kt < 4; ++kt) {
      int k0 = kt * 32;
      v16h a;
      // 16-bit A 16x32 layout: lo lanes K={k0..k0+7, k0+16..k0+23}, hi lanes +8
      ((float4*)&a)[0] = *(const float4*)(arowBase + k0);
      ((float4*)&a)[1] = *(const float4*)(arowBase + k0 + 16);

#pragma unroll
      for (int nt = 0; nt < 8; ++nt) {
        // B 32x16 layout: lane holds column n = nt*16+rlo; lo lanes K={k0..k0+7, k0+16..k0+23},
        // hi lanes +8 -> two ds_load_b128 from sW[n][k]
        const _Float16* bp = sW + (nt * 16 + rlo) * HID + k0 + hiHalf * 8;
        v16h b;
        ((float4*)&b)[0] = *(const float4*)(bp);
        ((float4*)&b)[1] = *(const float4*)(bp + 16);
        acc[nt] = __builtin_amdgcn_wmma_f32_16x16x32_f16(
            /*neg_a=*/false, a, /*neg_b=*/false, b,
            /*c_mod=*/(short)0, acc[nt], /*reuse_a=*/false, /*reuse_b=*/false);
      }
    }

    // C/D layout: VGPR v holds row M = v + 8*hiHalf, column N = rlo (per tile)
#pragma unroll
    for (int nt = 0; nt < 8; ++nt) {
#pragma unroll
      for (int v = 0; v < 8; ++v) {
        C[(size_t)(stripe * 16 + v + 8 * hiHalf) * HID + nt * 16 + rlo] = acc[nt][v];
      }
    }
  }
}

// ---------------------------------------------------------------------------
// Per-(node,head) attention dots: alpha_s/alpha_d = <h[n,head,:], a_s/a_d[head,:]>
// ---------------------------------------------------------------------------
__global__ void alpha_kernel(const float* __restrict__ h,
                             const float* __restrict__ a_s, const float* __restrict__ a_d,
                             float* __restrict__ als, float* __restrict__ ald, int nNodes) {
  int i = blockIdx.x * blockDim.x + threadIdx.x;   // i = node*HEADS + head
  if (i >= nNodes * HEADS) return;
  int head = i & (HEADS - 1);
  int node = i >> 2;
  const float* hp  = h + (size_t)node * HID + head * HEAD_DIM;
  const float* asp = a_s + head * HEAD_DIM;
  const float* adp = a_d + head * HEAD_DIM;
  float ss = 0.f, sd = 0.f;
#pragma unroll
  for (int c = 0; c < HEAD_DIM; ++c) {
    float v = hp[c];
    ss += v * asp[c];
    sd += v * adp[c];
  }
  als[i] = ss;
  ald[i] = sd;
}

// ---------------------------------------------------------------------------
// Edge pass 1: logits + LeakyReLU, store e, segment-max into smax (u32 atomics)
// ---------------------------------------------------------------------------
__global__ void edge_max(const int* __restrict__ src, const int* __restrict__ dst,
                         const float* __restrict__ als, const float* __restrict__ ald,
                         float* __restrict__ ews, unsigned* __restrict__ smax,
                         int nE, int nNodes) {
  int e = blockIdx.x * blockDim.x + threadIdx.x;
  if (e >= nE + nNodes) return;
  int s, d;
  if (e < nE) { s = src[e]; d = dst[e]; } else { s = d = e - nE; }
#pragma unroll
  for (int hh = 0; hh < HEADS; ++hh) {
    float v = als[s * HEADS + hh] + ald[d * HEADS + hh];
    v = v > 0.f ? v : NEG_SLOPE * v;             // LeakyReLU(0.2)
    ews[(size_t)e * HEADS + hh] = v;
    atomicMax(smax + d * HEADS + hh, enc_f32(v));
  }
}

// ---------------------------------------------------------------------------
// Edge pass 2: denom[d,h] += exp(e - segmax[d,h])
// ---------------------------------------------------------------------------
__global__ void edge_denom(const int* __restrict__ dst, const float* __restrict__ ews,
                           const unsigned* __restrict__ smax, float* __restrict__ denom,
                           int nE, int nNodes) {
  int e = blockIdx.x * blockDim.x + threadIdx.x;
  if (e >= nE + nNodes) return;
  int d = (e < nE) ? dst[e] : (e - nE);
#pragma unroll
  for (int hh = 0; hh < HEADS; ++hh) {
    float m = dec_f32(smax[d * HEADS + hh]);
    float p = __expf(ews[(size_t)e * HEADS + hh] - m);
    atomicAdd(denom + d * HEADS + hh, p);
  }
}

// ---------------------------------------------------------------------------
// Edge pass 3: agg[d,:] += alpha * h[src,:]  (one block = one edge, 128 ch)
// ---------------------------------------------------------------------------
__global__ void __launch_bounds__(128)
edge_agg(const int* __restrict__ src, const int* __restrict__ dst,
         const float* __restrict__ ews, const unsigned* __restrict__ smax,
         const float* __restrict__ denom, const float* __restrict__ h,
         float* __restrict__ agg, int nE) {
  int e = blockIdx.x;
  int t = threadIdx.x;                  // channel 0..127
  int s, d;
  if (e < nE) { s = src[e]; d = dst[e]; } else { s = d = e - nE; }
  int hh = t >> 5;                      // head
  float m     = dec_f32(smax[d * HEADS + hh]);
  float p     = __expf(ews[(size_t)e * HEADS + hh] - m);
  float alpha = p / denom[d * HEADS + hh];
  float val   = alpha * h[(size_t)s * HID + t];
  atomicAdd(agg + (size_t)d * HID + t, val);
}

// ---------------------------------------------------------------------------
// bias + ReLU; intermediate layers -> f16 activations for next WMMA GEMM,
// last layer -> f32 d_out
// ---------------------------------------------------------------------------
__global__ void gat_finalize(const float* __restrict__ agg, const float* __restrict__ bias,
                             _Float16* __restrict__ xh_next, float* __restrict__ out_final,
                             int n, int is_last) {
  int i = blockIdx.x * blockDim.x + threadIdx.x;
  if (i >= n) return;
  float v = agg[i] + bias[i & (HID - 1)];
  v = v > 0.f ? v : 0.f;
  if (is_last) out_final[i] = v;
  else         xh_next[i]   = (_Float16)v;
}

// ---------------------------------------------------------------------------
extern "C" void kernel_launch(void* const* d_in, const int* in_sizes, int n_in,
                              void* d_out, int out_size, void* d_ws, size_t ws_size,
                              hipStream_t stream) {
  const float* x      = (const float*)d_in[0];   // [N,128]
  const int*   eidx   = (const int*)d_in[1];     // [2,E] (jax x64 off -> int32)
  const float* WsA    = (const float*)d_in[2];   // [3,128,128]
  const float* a_srcA = (const float*)d_in[3];   // [3,4,32]
  const float* a_dstA = (const float*)d_in[4];   // [3,4,32]
  const float* biasA  = (const float*)d_in[5];   // [3,128]
  (void)n_in; (void)out_size; (void)ws_size;

  const int nNodes = in_sizes[0] / HID;          // 50000
  const int nE     = in_sizes[1] / 2;            // 800000
  const int nEtot  = nE + nNodes;                // + self loops
  const int* src   = eidx;
  const int* dst   = eidx + nE;

  // bump-allocate workspace
  char* ws = (char*)d_ws;
  size_t off = 0;
  auto alloc = [&](size_t bytes) -> void* {
    off = (off + 255) & ~(size_t)255;
    void* p = ws + off;
    off += bytes;
    return p;
  };
  _Float16* xh    = (_Float16*)alloc((size_t)nNodes * HID * sizeof(_Float16));
  float*    hbuf  = (float*)   alloc((size_t)nNodes * HID * sizeof(float));
  float*    als   = (float*)   alloc((size_t)nNodes * HEADS * sizeof(float));
  float*    ald   = (float*)   alloc((size_t)nNodes * HEADS * sizeof(float));
  unsigned* smax  = (unsigned*)alloc((size_t)nNodes * HEADS * sizeof(unsigned));
  float*    denom = (float*)   alloc((size_t)nNodes * HEADS * sizeof(float));
  float*    ews   = (float*)   alloc((size_t)nEtot * HEADS * sizeof(float));
  float*    agg   = (float*)   alloc((size_t)nNodes * HID * sizeof(float));

  const int nElems    = nNodes * HID;
  const int nRowTiles = nNodes / 16;             // 3125 (50000 % 16 == 0)

  // persistent-block GEMM grid: 160 blocks x 8 waves -> ~2-3 stripes per wave
  int gemmBlocks = (nRowTiles + 7) / 8;
  if (gemmBlocks > 160) gemmBlocks = 160;

  // initial activations -> f16
  cvt_x_f16<<<(nElems + 255) / 256, 256, 0, stream>>>(x, xh, nElems);

  for (int l = 0; l < 3; ++l) {
    // h = x @ W  (WMMA f16 -> f32, W staged in LDS)
    gat_gemm_wmma<<<gemmBlocks, 256, 0, stream>>>(
        xh, WsA + (size_t)l * HID * HID, hbuf, nRowTiles);

    // reset segment accumulators
    hipMemsetAsync(smax,  0, (size_t)nNodes * HEADS * sizeof(unsigned), stream);
    hipMemsetAsync(denom, 0, (size_t)nNodes * HEADS * sizeof(float),    stream);
    hipMemsetAsync(agg,   0, (size_t)nElems * sizeof(float),            stream);

    // attention scalars
    alpha_kernel<<<(nNodes * HEADS + 255) / 256, 256, 0, stream>>>(
        hbuf, a_srcA + l * HEADS * HEAD_DIM, a_dstA + l * HEADS * HEAD_DIM,
        als, ald, nNodes);

    // segment softmax + aggregation
    edge_max  <<<(nEtot + 255) / 256, 256, 0, stream>>>(src, dst, als, ald, ews, smax, nE, nNodes);
    edge_denom<<<(nEtot + 255) / 256, 256, 0, stream>>>(dst, ews, smax, denom, nE, nNodes);
    edge_agg  <<<nEtot, 128, 0, stream>>>(src, dst, ews, smax, denom, hbuf, agg, nE);

    // bias + ReLU (-> next-layer f16 activations, or final f32 output)
    gat_finalize<<<(nElems + 255) / 256, 256, 0, stream>>>(
        agg, biasA + l * HID, xh, (float*)d_out, nElems, l == 2);
  }
}
